// HGTCavAttention_86303072846495
// MI455X (gfx1250) — compile-verified, matrix operand order checked by reading, not run on hardware
//
#include <hip/hip_runtime.h>
#include <hip/hip_bf16.h>

// ---------------------------------------------------------------------------
// Problem constants (from reference)
// ---------------------------------------------------------------------------
#define B_      4
#define L_      5
#define HH      32
#define WW      32
#define HW      1024            // H*W
#define C_      256
#define MH      8               // heads
#define DH      32              // dim head
#define TH      2               // num types
#define RR      4               // num relations
#define INNER_  256             // M*D

typedef __bf16 bf16;
typedef __attribute__((ext_vector_type(16))) __bf16 v16bf;
typedef __attribute__((ext_vector_type(8)))  __bf16 v8bf;
typedef __attribute__((ext_vector_type(8)))  float  v8f;

// ---------------------------------------------------------------------------
// Fragment load for a K-major (row-major [rows][K]) bf16 matrix.
// Matches CDNA5 16-bit A/B WMMA VGPR layout:
//   lanes 0-15 : row = lane,     elems 0..7 -> K 0..7,  elems 8..15 -> K 16..23
//   lanes16-31 : row = lane-16,  elems 0..7 -> K 8..15, elems 8..15 -> K 24..31
// ---------------------------------------------------------------------------
__device__ __forceinline__ v16bf frag_ld(const bf16* __restrict__ p,
                                         long rstride, int lane) {
  const int r = lane & 15;
  const int c = (lane >> 4) << 3;     // 0 or 8
  union { v16bf v; v8bf h[2]; } u;
  u.h[0] = *(const v8bf*)(p + (long)r * rstride + c);
  u.h[1] = *(const v8bf*)(p + (long)r * rstride + c + 16);
  return u.v;
}

__device__ __forceinline__ v8f wmma_bf16(v16bf a, v16bf b, v8f c) {
  // D = A(16x32 bf16) * B(32x16 bf16) + C(16x16 f32)
  return __builtin_amdgcn_wmma_f32_16x16x32_bf16(
      /*neg_a=*/false, a, /*neg_b=*/false, b,
      /*c_mod=*/(short)0, c, /*reuse_a=*/false, /*reuse_b=*/false);
}

// ---------------------------------------------------------------------------
// K0: extract per-(b,l) type from prior_encoding[:, :, 0, 0, 2]
// ---------------------------------------------------------------------------
__global__ void __launch_bounds__(32)
prep_types_kernel(const float* __restrict__ pe, int* __restrict__ types) {
  int i = threadIdx.x;
  if (i < B_ * L_) {
    types[i] = (int)pe[(long)i * HW * 3 + 2];
  }
}

// ---------------------------------------------------------------------------
// K1: f32 -> bf16 conversion (straight copy)
// ---------------------------------------------------------------------------
__global__ void __launch_bounds__(256)
cvt_bf16_kernel(const float* __restrict__ src, bf16* __restrict__ dst, int n) {
  int i = blockIdx.x * 256 + threadIdx.x;
  if (i < n) dst[i] = (bf16)src[i];
}

// relation_msg [r][m][p][c] -> wmsgT [r][m][c][p]  (so msg GEMM B is K-major)
__global__ void __launch_bounds__(256)
cvt_msgT_kernel(const float* __restrict__ src, bf16* __restrict__ dst) {
  int i = blockIdx.x * 256 + threadIdx.x;
  if (i < RR * MH * DH * DH) {
    int c  = i % DH;
    int p  = (i / DH) % DH;
    int rm = i / (DH * DH);
    dst[(long)rm * DH * DH + (long)c * DH + p] = (bf16)src[i];
  }
}

// ---------------------------------------------------------------------------
// K2: QKV projections.  q[n,o] = sum_c x[n,c] * W[o,c] + b[o]  (per-type W)
// grid.x = B*L*64 (hw 16-tiles); block = 256 (8 waves).
// Each wave: 2 o-tiles per matrix (q,k,v unrolled at compile time so every
// load uses a direct parameter pointer -> global_load, never flat_load).
// Output [b][m][l][hw][d] bf16.
// ---------------------------------------------------------------------------
__global__ void __launch_bounds__(256)
gemm_qkv_kernel(const bf16* __restrict__ xbf,
                const bf16* __restrict__ wq, const bf16* __restrict__ wk,
                const bf16* __restrict__ wv,
                const float* __restrict__ bq, const float* __restrict__ bk,
                const float* __restrict__ bv,
                const int* __restrict__ types,
                bf16* __restrict__ qo, bf16* __restrict__ ko,
                bf16* __restrict__ vo) {
  const int lane = threadIdx.x & 31;
  const int wave = threadIdx.x >> 5;
  const int blk  = blockIdx.x;
  const int nt   = blk & 63;              // hw 16-tile
  const int bl   = blk >> 6;              // b*L + l
  const int b    = bl / L_;
  const int l    = bl % L_;
  const int t    = types[bl];

  const bf16* xa = xbf + ((long)bl * HW + (long)nt * 16) * C_;

  // o-tiles per matrix handled by this wave: o16 = wave*2 + u, u in {0,1}
  v8f accq[2], acck[2], accv[2];
  for (int u = 0; u < 2; ++u)
    for (int r = 0; r < 8; ++r) { accq[u][r] = 0.f; acck[u][r] = 0.f; accv[u][r] = 0.f; }

  const long wbase = (long)t * INNER_ * C_ + (long)(wave * 2) * 16 * C_;

  for (int ks = 0; ks < C_; ks += 32) {
    v16bf afrag = frag_ld(xa + ks, C_, lane);
#pragma unroll
    for (int u = 0; u < 2; ++u) {
      const long wo = wbase + (long)u * 16 * C_ + ks;
      accq[u] = wmma_bf16(afrag, frag_ld(wq + wo, C_, lane), accq[u]);
      acck[u] = wmma_bf16(afrag, frag_ld(wk + wo, C_, lane), acck[u]);
      accv[u] = wmma_bf16(afrag, frag_ld(wv + wo, C_, lane), accv[u]);
    }
  }

  const int col = lane & 15;
  const int rhi = (lane >> 4) << 3;       // 0 or 8
#pragma unroll
  for (int u = 0; u < 2; ++u) {
    const int o16   = wave * 2 + u;
    const int m     = o16 >> 1;
    const int dbase = (o16 & 1) * 16;
    const int o     = m * DH + dbase + col;
    const long obase = ((((long)b * MH + m) * L_ + l) * HW) * DH;

    const float bbq = bq[t * INNER_ + o];
    const float bbk = bk[t * INNER_ + o];
    const float bbv = bv[t * INNER_ + o];
#pragma unroll
    for (int r = 0; r < 8; ++r) {
      const int hw = nt * 16 + r + rhi;
      const long off = obase + (long)hw * DH + dbase + col;
      qo[off] = (bf16)(accq[u][r] + bbq);
      ko[off] = (bf16)(acck[u][r] + bbk);
      vo[off] = (bf16)(accv[u][r] + bbv);
    }
  }
}

// ---------------------------------------------------------------------------
// K3: relation attention. grid.x = B*M*8, block = 256 (8 waves);
// wave -> one (b, m, hw 16-tile). Per wave:
//   scores: t = k_j * Watt(rel)^T via WMMA, dot with q rows, softmax over j,
//   output: msg = v_j * WmsgT(rel), out_i += att[i][j] * msg (fp32),
// stored as bf16 [b][l][hw][inner].
// ---------------------------------------------------------------------------
__global__ void __launch_bounds__(256)
attn_kernel(const bf16* __restrict__ qb, const bf16* __restrict__ kb,
            const bf16* __restrict__ vb,
            const bf16* __restrict__ watt,    // [r][m][p][q]
            const bf16* __restrict__ wmsgT,   // [r][m][c][p]
            const float* __restrict__ mask,   // [b][hw][i][j]
            const int* __restrict__ types,
            bf16* __restrict__ attnout) {     // [b][l][hw][inner]
  __shared__ float t_lds[8][16][32];
  __shared__ float a_lds[8][L_][L_][16];

  const int lane = threadIdx.x & 31;
  const int wave = threadIdx.x >> 5;
  const int blk  = blockIdx.x;
  const int tg   = blk & 7;
  const int bm   = blk >> 3;
  const int b    = bm / MH;
  const int m    = bm % MH;
  const int ht   = tg * 8 + wave;         // hw 16-tile, 0..63
  const int hw0  = ht * 16;

  int tb[L_];
  for (int j = 0; j < L_; ++j) tb[j] = types[b * L_ + j];

  // K and V fragments for all j (A operand: rows=hw, K=d)
  v16bf kf[L_], vf[L_];
  for (int j = 0; j < L_; ++j) {
    const bf16* kp = kb + ((((long)b * MH + m) * L_ + j) * HW + hw0) * DH;
    const bf16* vp = vb + ((((long)b * MH + m) * L_ + j) * HW + hw0) * DH;
    kf[j] = frag_ld(kp, DH, lane);
    vf[j] = frag_ld(vp, DH, lane);
  }

  // Q values for the score dot: lane owns (row = lane&15, p-half = lane>>4)
  float qv[L_][16];
  for (int i = 0; i < L_; ++i) {
    const bf16* qp = qb + ((((long)b * MH + m) * L_ + i) * HW + hw0 + (lane & 15)) * DH
                     + (long)(lane >> 4) * 16;
    for (int p = 0; p < 16; ++p) qv[i][p] = (float)qp[p];
  }

  const float scale = 0.17677669529663687f;   // 1/sqrt(32)
  const int row  = lane & 15;
  const int half = lane >> 4;
  const int rhi  = half << 3;

  // ---- scores ----
  for (int j = 0; j < L_; ++j) {
    for (int i = 0; i < L_; ++i) {
      int rel = tb[i] * TH + tb[j];
      const bf16* wp = watt + (((long)rel * MH + m) * DH) * DH;   // [p][q]
#pragma unroll
      for (int ph = 0; ph < 2; ++ph) {
        v8f z;
        for (int r = 0; r < 8; ++r) z[r] = 0.f;
        v8f d = wmma_bf16(kf[j], frag_ld(wp + (long)ph * 16 * DH, DH, lane), z);
        int colp = ph * 16 + (lane & 15);
        for (int r = 0; r < 8; ++r) t_lds[wave][r + rhi][colp] = d[r];
      }
      float s = 0.f;
      for (int p = 0; p < 16; ++p) s += qv[i][p] * t_lds[wave][row][half * 16 + p];
      s += __shfl_xor(s, 16, 32);
      if (lane < 16) a_lds[wave][i][j][row] = s * scale;
    }
  }

  // ---- masked softmax over j (per i, per row) ----
  for (int i = 0; i < L_; ++i) {
    if (lane < 16) {
      int hw = hw0 + row;
      const float* mp = mask + (((long)b * HW + hw) * L_ + i) * L_;
      float v[L_];
      float mx = -1e30f;
      for (int j = 0; j < L_; ++j) {
        float a = a_lds[wave][i][j][row];
        a = (mp[j] == 0.f) ? -1e30f : a;
        v[j] = a;
        mx = fmaxf(mx, a);
      }
      float sum = 0.f;
      for (int j = 0; j < L_; ++j) { v[j] = __expf(v[j] - mx); sum += v[j]; }
      float inv = 1.f / sum;
      for (int j = 0; j < L_; ++j) a_lds[wave][i][j][row] = v[j] * inv;
    }
  }

  // ---- weighted message aggregation ----
  const int col = lane & 15;
  for (int i = 0; i < L_; ++i) {
    v8f acc0, acc1;
    for (int r = 0; r < 8; ++r) { acc0[r] = 0.f; acc1[r] = 0.f; }
    for (int j = 0; j < L_; ++j) {
      int rel = tb[i] * TH + tb[j];
      const bf16* wp = wmsgT + (((long)rel * MH + m) * DH) * DH;  // [c][p]
      v8f z;
      for (int r = 0; r < 8; ++r) z[r] = 0.f;
      v8f m0 = wmma_bf16(vf[j], frag_ld(wp, DH, lane), z);
      v8f m1 = wmma_bf16(vf[j], frag_ld(wp + 16 * DH, DH, lane), z);
      float aw[8];
      for (int r = 0; r < 8; ++r) aw[r] = a_lds[wave][i][j][r + rhi];
      for (int r = 0; r < 8; ++r) { acc0[r] += aw[r] * m0[r]; acc1[r] += aw[r] * m1[r]; }
    }
    bf16* op = attnout + (((long)b * L_ + i) * HW) * INNER_ + (long)m * DH;
    for (int r = 0; r < 8; ++r) {
      int hw = hw0 + r + rhi;
      op[(long)hw * INNER_ + col]      = (bf16)acc0[r];
      op[(long)hw * INNER_ + 16 + col] = (bf16)acc1[r];
    }
  }
}

// ---------------------------------------------------------------------------
// K4: output projection. out[n,o] = sum_in a[n,in] * Wa[o,in] + ba[o]
// grid.x = B*L*64; block = 256 (8 waves); wave handles 2 o-tiles of 16.
// Writes final f32 output directly in (b,l,h,w,c) layout.
// ---------------------------------------------------------------------------
__global__ void __launch_bounds__(256)
gemm_out_kernel(const bf16* __restrict__ ain,   // [b][l][hw][inner]
                const bf16* __restrict__ wa,    // [t][o][inner]
                const float* __restrict__ ba,   // [t][o]
                const int* __restrict__ types,
                float* __restrict__ out) {      // [b][l][hw][c]
  const int lane = threadIdx.x & 31;
  const int wave = threadIdx.x >> 5;
  const int blk  = blockIdx.x;
  const int nt   = blk & 63;
  const int bl   = blk >> 6;
  const int t    = types[bl];

  const bf16* xa = ain + ((long)bl * HW + (long)nt * 16) * INNER_;

  v8f acc[2];
  for (int u = 0; u < 2; ++u)
    for (int r = 0; r < 8; ++r) acc[u][r] = 0.f;

  for (int ks = 0; ks < INNER_; ks += 32) {
    v16bf afrag = frag_ld(xa + ks, INNER_, lane);
#pragma unroll
    for (int u = 0; u < 2; ++u) {
      int o16 = wave * 2 + u;
      const bf16* wb = wa + ((long)t * C_ + (long)o16 * 16) * INNER_ + ks;
      acc[u] = wmma_bf16(afrag, frag_ld(wb, INNER_, lane), acc[u]);
    }
  }

  const int col = lane & 15;
  const int rhi = (lane >> 4) << 3;
#pragma unroll
  for (int u = 0; u < 2; ++u) {
    int o    = (wave * 2 + u) * 16 + col;
    float bb = ba[t * C_ + o];
    for (int r = 0; r < 8; ++r) {
      int hw = nt * 16 + r + rhi;
      out[((long)bl * HW + hw) * C_ + o] = acc[u][r] + bb;
    }
  }
}

// ---------------------------------------------------------------------------
// Host launcher
// ---------------------------------------------------------------------------
extern "C" void kernel_launch(void* const* d_in, const int* in_sizes, int n_in,
                              void* d_out, int out_size, void* d_ws, size_t ws_size,
                              hipStream_t stream) {
  (void)in_sizes; (void)n_in; (void)out_size; (void)ws_size;

  const float* x    = (const float*)d_in[0];   // (B,L,H,W,C)
  const float* mask = (const float*)d_in[1];   // (B,H,W,L,L)
  const float* pe   = (const float*)d_in[2];   // (B,L,H,W,3)
  const float* q_w  = (const float*)d_in[3];   // (T,INNER,C)
  const float* q_b  = (const float*)d_in[4];   // (T,INNER)
  const float* k_w  = (const float*)d_in[5];
  const float* k_b  = (const float*)d_in[6];
  const float* v_w  = (const float*)d_in[7];
  const float* v_b  = (const float*)d_in[8];
  const float* a_w  = (const float*)d_in[9];   // (T,C,INNER)
  const float* a_b  = (const float*)d_in[10];  // (T,C)
  const float* ratt = (const float*)d_in[11];  // (R,M,D,D)
  const float* rmsg = (const float*)d_in[12];  // (R,M,D,D)
  float* out = (float*)d_out;                  // (B,L,H,W,C)

  // ---- workspace carve-up (256B aligned) ----
  char* w = (char*)d_ws;
  size_t off = 0;
  auto carve = [&](size_t bytes) -> void* {
    void* p = w + off;
    off = (off + bytes + 255) & ~(size_t)255;
    return p;
  };
  const size_t N_X   = (size_t)B_ * L_ * HW * C_;           // 5,242,880
  const size_t N_W   = (size_t)TH * INNER_ * C_;            // 131,072
  const size_t N_REL = (size_t)RR * MH * DH * DH;           // 32,768
  const size_t N_QKV = (size_t)B_ * MH * L_ * HW * DH;      // 5,242,880

  int*  types   = (int*)carve(B_ * L_ * sizeof(int));
  bf16* x_bf    = (bf16*)carve(N_X * 2);
  bf16* qw_bf   = (bf16*)carve(N_W * 2);
  bf16* kw_bf   = (bf16*)carve(N_W * 2);
  bf16* vw_bf   = (bf16*)carve(N_W * 2);
  bf16* aw_bf   = (bf16*)carve(N_W * 2);
  bf16* watt_bf = (bf16*)carve(N_REL * 2);
  bf16* wmsgT   = (bf16*)carve(N_REL * 2);
  bf16* q_bf    = (bf16*)carve(N_QKV * 2);
  bf16* k_bf    = (bf16*)carve(N_QKV * 2);
  bf16* v_bf    = (bf16*)carve(N_QKV * 2);
  bf16* attnout = (bf16*)carve(N_X * 2);

  // ---- stage 0: types ----
  prep_types_kernel<<<1, 32, 0, stream>>>(pe, types);

  // ---- stage 1: conversions ----
  auto cvt = [&](const float* s, bf16* d, size_t n) {
    cvt_bf16_kernel<<<(int)((n + 255) / 256), 256, 0, stream>>>(s, d, (int)n);
  };
  cvt(x, x_bf, N_X);
  cvt(q_w, qw_bf, N_W);
  cvt(k_w, kw_bf, N_W);
  cvt(v_w, vw_bf, N_W);
  cvt(a_w, aw_bf, N_W);
  cvt(ratt, watt_bf, N_REL);
  cvt_msgT_kernel<<<(int)((N_REL + 255) / 256), 256, 0, stream>>>(rmsg, wmsgT);

  // ---- stage 2: QKV projection (WMMA) ----
  gemm_qkv_kernel<<<B_ * L_ * 64, 256, 0, stream>>>(
      x_bf, qw_bf, kw_bf, vw_bf, q_b, k_b, v_b, types, q_bf, k_bf, v_bf);

  // ---- stage 3: relation attention (WMMA) ----
  attn_kernel<<<B_ * MH * 8, 256, 0, stream>>>(
      q_bf, k_bf, v_bf, watt_bf, wmsgT, mask, types, attnout);

  // ---- stage 4: output projection (WMMA) ----
  gemm_out_kernel<<<B_ * L_ * 64, 256, 0, stream>>>(
      attnout, aw_bf, a_b, types, out);
}
